// MetaPath_67757404062506
// MI455X (gfx1250) — matrix operand order repaired and checked by reading, not compiled, not run on hardware
//
#include <hip/hip_runtime.h>
#include <float.h>

typedef __bf16 v16bf __attribute__((ext_vector_type(16)));
typedef float  v8f   __attribute__((ext_vector_type(8)));
using gv16bf = __attribute__((address_space(1))) v16bf;   // global-AS fragment

#define DF    64     // feature dim
#define NH    8      // heads
#define HID   512    // H*D
#define NG    2048   // 4*HID (gates)
#define KCAT  576    // DF + HID (concat GEMM K)
#define KT    18     // KCAT/32 k-tiles
#define NT    128    // NG/16 n-tiles
#define CATP  584    // padded LDS row stride (bf16 elems)

__device__ __forceinline__ __bf16 f2bf(float f) {
  union { unsigned short s; __bf16 b; } u;
  unsigned x = __float_as_uint(f);
  u.s = (unsigned short)((x + 0x7FFFu + ((x >> 16) & 1u)) >> 16);  // RTNE
  return u.b;
}
__device__ __forceinline__ unsigned f2ord(float f) {   // order-preserving f32->u32
  unsigned u = __float_as_uint(f);
  return (u & 0x80000000u) ? ~u : (u | 0x80000000u);
}
__device__ __forceinline__ float ord2f(unsigned v) {
  unsigned u = (v & 0x80000000u) ? (v & 0x7FFFFFFFu) : ~v;
  return __uint_as_float(u);
}
__device__ __forceinline__ float sigm(float x) { return 1.0f / (1.0f + __expf(-x)); }

// ---------------------------------------------------------------------------
// Pack [W_ih^T ; W_hh^T] (KCAT x NG) into bf16 WMMA B-fragments:
// Bfrag[nt][kt][lane][i], element i of lane: K = kt*32 + i + (lane>=16 ? 16:0),
// N = nt*16 + (lane&15)   (contiguous k order, same as the A fragments).
// ---------------------------------------------------------------------------
__global__ __launch_bounds__(256) void pack_b_kernel(
    const float* __restrict__ Wih, const float* __restrict__ Whh,
    __bf16* __restrict__ Bfrag)
{
  int tid = blockIdx.x * 256 + threadIdx.x;        // NT*KT*32*16 threads exactly
  int i    = tid & 15;
  int lane = (tid >> 4) & 31;
  int kt   = (tid >> 9) % KT;
  int nt   = tid / (512 * KT);
  int K = kt * 32 + i + ((lane & 16) ? 16 : 0);
  int N = nt * 16 + (lane & 15);
  float v = (K < DF) ? Wih[(size_t)N * DF + K] : Whh[(size_t)N * HID + (K - DF)];
  Bfrag[tid] = f2bf(v);
}

__global__ __launch_bounds__(256) void pack_bias_kernel(
    const float* __restrict__ bih, const float* __restrict__ bhh,
    float* __restrict__ bsum)
{
  int t = blockIdx.x * 256 + threadIdx.x;
  if (t < NG) bsum[t] = bih[t] + bhh[t];
}

__global__ __launch_bounds__(256) void init_out_kernel(float* __restrict__ out, int n) {
  int t = blockIdx.x * 256 + threadIdx.x;
  if (t < n) out[t] = 0.0f;
}
__global__ __launch_bounds__(256) void init_nh_kernel(unsigned* __restrict__ amaxU,
                                                      float* __restrict__ denom, int n) {
  int t = blockIdx.x * 256 + threadIdx.x;
  if (t < n) { amaxU[t] = 0u; denom[t] = 0.0f; }  // ord(0)=lowest; only read where edges exist
}

// ---------------------------------------------------------------------------
// Fused 4-step LSTM over metapath instances, 16 edges per workgroup.
// gates(16,2048) = [x|h](16,576) @ Wcat(576,2048) via v_wmma_f32_16x16x32_bf16.
// h (bf16, LDS) + c (registers) never leave the WGP until the final step.
// B fragments stream from L2 each step via global_load_b128; an opaque-pointer
// barrier per (tile,gate) group blocks LICM so they are never spilled.
// ---------------------------------------------------------------------------
__global__ __launch_bounds__(256) void lstm_kernel(
    const float* __restrict__ features, const int* __restrict__ eidx,
    const __bf16* __restrict__ Bfrag, const float* __restrict__ bsum,
    float* __restrict__ eft, int E, int L)
{
  __shared__ alignas(32) __bf16 catA[16 * CATP];      // [x(0:64) | h(64:576)] rows, bf16
  __shared__ alignas(32) __bf16 Afrag[KT * 32 * 16];  // A fragments in WMMA lane layout

  const int tid  = threadIdx.x;
  const int lane = tid & 31;
  const int wave = tid >> 5;
  const int e0   = blockIdx.x * 16;

  for (int q = tid; q < 16 * CATP; q += 256) catA[q] = f2bf(0.0f);  // h0 = 0

  v8f creg[4];                                        // c state: 4 hidden tiles / wave
  #pragma unroll
  for (int q = 0; q < 4; ++q)
    creg[q] = v8f{0.f,0.f,0.f,0.f,0.f,0.f,0.f,0.f};

  __syncthreads();

  const v16bf* Ap = reinterpret_cast<const v16bf*>(Afrag);
  const int col = lane & 15;             // N within tile (C/D layout)
  const int hi  = (lane >> 4) & 1;       // row-half selector (C/D layout)

  for (int t = 0; t < L; ++t) {
    // 1) gather x_t rows -> catA[:, 0:64]   (16 threads x 4 floats per row)
    {
      int r  = tid >> 4;
      int cg = tid & 15;
      int e  = e0 + r;
      int ec = (e < E) ? e : (E - 1);
      int node = eidx[ec * L + t];
      const float4 v = *reinterpret_cast<const float4*>(features + (size_t)node * DF + cg * 4);
      catA[r * CATP + cg * 4 + 0] = f2bf(v.x);
      catA[r * CATP + cg * 4 + 1] = f2bf(v.y);
      catA[r * CATP + cg * 4 + 2] = f2bf(v.z);
      catA[r * CATP + cg * 4 + 3] = f2bf(v.w);
    }
    __syncthreads();   // x_t and previous-step h visible to all waves

    // 2) build A fragments (contiguous k order, as in Bfrag: K = i + 16*(lane>=16))
    #pragma unroll
    for (int s = 0; s < 36; ++s) {       // 36*256 == KT*32*16
      int q   = tid + (s << 8);
      int kt  = q >> 9;
      int rem = q & 511;
      int l   = rem >> 4;
      int i   = rem & 15;
      int M   = l & 15;
      int K   = kt * 32 + i + ((l & 16) ? 16 : 0);
      Afrag[q] = catA[M * CATP + K];
    }
    __syncthreads();

    // 3) preload all K-fragments of A into VGPRs (shared by all 16 output tiles)
    v16bf areg[KT];
    #pragma unroll
    for (int kt = 0; kt < KT; ++kt) areg[kt] = Ap[kt * 32 + lane];

    // 4) each wave: 4 hidden tiles x 4 gates x 18 WMMAs
    #pragma unroll
    for (int q = 0; q < 4; ++q) {
      const int ht = wave * 4 + q;
      v8f acc[4];
      #pragma unroll
      for (int g = 0; g < 4; ++g)
        acc[g] = v8f{0.f,0.f,0.f,0.f,0.f,0.f,0.f,0.f};
      #pragma unroll
      for (int g = 0; g < 4; ++g) {
        const int nt = g * 32 + ht;
        // Global-AS pointer so loads lower to global_load_b128 (LOADcnt only,
        // no flat/DScnt coupling); opaque barrier defeats LICM across t.
        const gv16bf* bb = (const gv16bf*)(
            reinterpret_cast<const v16bf*>(Bfrag) + (size_t)nt * (KT * 32) + lane);
        asm volatile("" : "+v"(bb));
        // 3-deep software pipeline on the weight stream (hides L2 latency)
        v16bf b0 = bb[0];
        v16bf b1 = bb[32];
        v16bf b2 = bb[64];
        #pragma unroll
        for (int kt = 0; kt < KT; ++kt) {
          v16bf bn = (kt + 3 < KT) ? bb[(kt + 3) * 32] : b0;
          acc[g] = __builtin_amdgcn_wmma_f32_16x16x32_bf16(
              false, areg[kt], false, b0, (short)0, acc[g], false, false);
          b0 = b1; b1 = b2; b2 = bn;
        }
      }
      // 5) LSTM cell on this 16x16 hidden tile (acc/creg share the C/D layout)
      const int n  = ht * 16 + col;
      const float bi = bsum[0 * HID + n];
      const float bf = bsum[1 * HID + n];
      const float bg = bsum[2 * HID + n];
      const float bo = bsum[3 * HID + n];
      #pragma unroll
      for (int r = 0; r < 8; ++r) {
        const int M = r + hi * 8;
        float gi = sigm(acc[0][r] + bi);
        float gf = sigm(acc[1][r] + bf);
        float gg = tanhf(acc[2][r] + bg);
        float go = sigm(acc[3][r] + bo);
        float cn = gf * creg[q][r] + gi * gg;
        float hn = go * tanhf(cn);
        creg[q][r] = cn;
        catA[M * CATP + DF + n] = f2bf(hn);          // feed next timestep
        if (t == L - 1) {
          int e = e0 + M;
          if (e < E) eft[(size_t)e * HID + n] = hn;  // full-precision final hidden
        }
      }
    }
    __syncthreads();
  }
}

// ---------------------------------------------------------------------------
// a = leaky_relu(center@attn1^T + sum(eft*attn2)) ; segment max via ordered u32
// one wave per edge: 8 heads x 4 lanes, 16 elems/lane, butterfly reduce.
// ---------------------------------------------------------------------------
__global__ __launch_bounds__(256) void attn_a_kernel(
    const float* __restrict__ features, const int* __restrict__ eidx,
    const float* __restrict__ attn1_w, const float* __restrict__ attn2,
    const float* __restrict__ eft, const int* __restrict__ dst,
    float* __restrict__ aArr, unsigned* __restrict__ amaxU, int E, int L)
{
  const int lane = threadIdx.x & 31;
  const int e = blockIdx.x * 8 + (threadIdx.x >> 5);
  if (e >= E) return;
  const int head = lane >> 2;
  const int sub  = lane & 3;
  const int node = eidx[e * L + (L - 1)];
  float s = 0.0f;
  #pragma unroll
  for (int j = 0; j < 16; ++j) {
    int d = sub * 16 + j;
    s += features[(size_t)node * DF + d] * attn1_w[head * DF + d];
    s += eft[(size_t)e * HID + head * DF + d] * attn2[head * DF + d];
  }
  s += __shfl_xor(s, 1, 32);
  s += __shfl_xor(s, 2, 32);
  float a = (s > 0.0f) ? s : 0.01f * s;
  if (sub == 0) {
    aArr[e * NH + head] = a;
    atomicMax(&amaxU[(size_t)dst[e] * NH + head], f2ord(a));
  }
}

__global__ __launch_bounds__(256) void exp_kernel(
    const int* __restrict__ dst, float* __restrict__ aArr,
    const unsigned* __restrict__ amaxU, float* __restrict__ denom, int EH)
{
  int tid = blockIdx.x * 256 + threadIdx.x;
  if (tid >= EH) return;
  int e = tid >> 3, h = tid & 7;
  int d = dst[e];
  float m  = ord2f(amaxU[(size_t)d * NH + h]);
  float ex = __expf(aArr[tid] - m);
  aArr[tid] = ex;
  atomicAdd(&denom[(size_t)d * NH + h], ex);
}

__global__ __launch_bounds__(256) void agg_kernel(
    const int* __restrict__ dst, const float* __restrict__ aArr,
    const float* __restrict__ denom, const float* __restrict__ eft,
    float* __restrict__ out, int E)
{
  const int lane = threadIdx.x & 31;
  const int e = blockIdx.x * 8 + (threadIdx.x >> 5);
  if (e >= E) return;
  const int d = dst[e];
  #pragma unroll
  for (int m = 0; m < NH; ++m) {
    float attn = aArr[e * NH + m] / denom[(size_t)d * NH + m];
    int j = m * DF + lane;
    atomicAdd(&out[(size_t)d * HID + j],      eft[(size_t)e * HID + j]      * attn);
    atomicAdd(&out[(size_t)d * HID + j + 32], eft[(size_t)e * HID + j + 32] * attn);
  }
}

// ---------------------------------------------------------------------------
extern "C" void kernel_launch(void* const* d_in, const int* in_sizes, int n_in,
                              void* d_out, int out_size, void* d_ws, size_t ws_size,
                              hipStream_t stream)
{
  (void)n_in; (void)ws_size; (void)out_size;
  const float* features = (const float*)d_in[0];
  // d_in[1] = type_mask: unused by the reference
  const int*   eidx     = (const int*)d_in[2];
  const int*   dst      = (const int*)d_in[3];
  const float* W_ih     = (const float*)d_in[4];
  const float* W_hh     = (const float*)d_in[5];
  const float* b_ih     = (const float*)d_in[6];
  const float* b_hh     = (const float*)d_in[7];
  const float* attn1_w  = (const float*)d_in[8];
  const float* attn2    = (const float*)d_in[9];
  float* out = (float*)d_out;

  const int N = in_sizes[0] / DF;
  const int E = in_sizes[3];
  const int L = in_sizes[2] / E;

  // workspace layout (256B-aligned sections)
  char* ws = (char*)d_ws;
  size_t off = 0;
  auto take = [&](size_t bytes) -> char* {
    char* p = ws + off;
    off = (off + bytes + 255) & ~(size_t)255;
    return p;
  };
  __bf16*   Bfrag = (__bf16*)  take((size_t)NT * KT * 32 * 16 * sizeof(__bf16));
  float*    bsum  = (float*)   take((size_t)NG * sizeof(float));
  float*    eft   = (float*)   take((size_t)E * HID * sizeof(float));
  float*    aArr  = (float*)   take((size_t)E * NH * sizeof(float));
  unsigned* amaxU = (unsigned*)take((size_t)N * NH * sizeof(unsigned));
  float*    denom = (float*)   take((size_t)N * NH * sizeof(float));

  // weight/bias packing + output/stat init (independent, stream-ordered)
  pack_b_kernel<<<(NT * KT * 32 * 16) / 256, 256, 0, stream>>>(W_ih, W_hh, Bfrag);
  pack_bias_kernel<<<(NG + 255) / 256, 256, 0, stream>>>(b_ih, b_hh, bsum);
  init_out_kernel<<<((size_t)N * HID + 255) / 256, 256, 0, stream>>>(out, N * HID);
  init_nh_kernel<<<(N * NH + 255) / 256, 256, 0, stream>>>(amaxU, denom, N * NH);

  // fused WMMA LSTM over all metapath instances
  lstm_kernel<<<(E + 15) / 16, 256, 0, stream>>>(features, eidx, Bfrag, bsum, eft, E, L);

  // attention: logits + segment max, exp + segment sum, weighted aggregation
  attn_a_kernel<<<(E + 7) / 8, 256, 0, stream>>>(features, eidx, attn1_w, attn2,
                                                 eft, dst, aArr, amaxU, E, L);
  exp_kernel<<<(E * NH + 255) / 256, 256, 0, stream>>>(dst, aArr, amaxU, denom, E * NH);
  agg_kernel<<<(E + 7) / 8, 256, 0, stream>>>(dst, aArr, denom, eft, out, E);
}
